// SelfAttention_88441966559986
// MI455X (gfx1250) — matrix operand order repaired and checked by reading, not compiled, not run on hardware
//
#include <hip/hip_runtime.h>
#include <math.h>

typedef _Float16 v16h __attribute__((ext_vector_type(16)));
typedef _Float16 v8h  __attribute__((ext_vector_type(8)));
typedef _Float16 v4h  __attribute__((ext_vector_type(4)));
typedef float    v8f  __attribute__((ext_vector_type(8)));
typedef int      v4i  __attribute__((ext_vector_type(4)));

#define D_MODEL 1024
#define N_HEADS 16
#define D_HEAD  64
#define SEQ_T   2048
#define BATCH   2
#define NTOK    (BATCH * SEQ_T)   // 4096

// ---- CDNA5 async global->LDS (ASYNCcnt) with fallback ----
#if defined(__has_builtin)
# if __has_builtin(__builtin_amdgcn_global_load_async_to_lds_b128)
#  define HAVE_ASYNC_LDS 1
# endif
#endif
#ifndef HAVE_ASYNC_LDS
# define HAVE_ASYNC_LDS 0
#endif

typedef __attribute__((address_space(1))) v4i* gv4i_ptr;
typedef __attribute__((address_space(3))) v4i* lv4i_ptr;

__device__ __forceinline__ void async_copy16(const _Float16* g, _Float16* l) {
#if HAVE_ASYNC_LDS
    __builtin_amdgcn_global_load_async_to_lds_b128(
        (gv4i_ptr)(void*)g,
        (lv4i_ptr)(void*)l, 0, 0);
#else
    *(v8h*)l = *(const v8h*)g;
#endif
}
__device__ __forceinline__ void wait_async_all() {
#if HAVE_ASYNC_LDS
    asm volatile("s_wait_asynccnt 0" ::: "memory");
#endif
}

// ---- WMMA fragment loaders (16-bit, wave32; same K-contiguous order for A and B) ----
// A 16x32 (MxK): lane L holds row M=L&15; K = (L>=16?16:0)+{0..15} contiguous.
__device__ __forceinline__ v16h load_a_frag(const _Float16* __restrict__ base, int ldk, int lane) {
    const int m  = lane & 15;
    const int kb = (lane >> 4) * 16;
    const _Float16* p = base + (size_t)m * ldk + kb;
    union { v16h v; v8h h[2]; } u;
    u.h[0] = *(const v8h*)(p);
    u.h[1] = *(const v8h*)(p + 8);
    return u.v;
}

// B 32x16 (KxN) read from B^T stored row-major [N][K]: lane L holds col N=L&15,
// K = (L>=16?16:0)+{0..15} contiguous -> one 32B load.
__device__ __forceinline__ v16h load_b_frag(const _Float16* __restrict__ baseT, int ldk, int lane) {
    const int n  = lane & 15;
    const int kb = (lane >> 4) * 16;
    return *(const v16h*)(baseT + (size_t)n * ldk + kb);
}

// B fragment from LDS-staged chunk: Bs[64 rows][stride 40 halves], K-chunk = 32
__device__ __forceinline__ v16h lds_b_frag(const _Float16* base, int j, int lane) {
    const int n  = j * 16 + (lane & 15);
    const int kb = (lane >> 4) * 16;
    const _Float16* p = base + n * 40 + kb;
    union { v16h v; v8h h[2]; } u;
    u.h[0] = *(const v8h*)(p);
    u.h[1] = *(const v8h*)(p + 8);
    return u.v;
}

// Cooperative stage of one 64(N) x 32(K) f16 B-chunk into LDS: 256 thr x 16B
__device__ __forceinline__ void stage_b_chunk(const _Float16* __restrict__ gsrc,
                                              _Float16* lbuf, int tid) {
    const int row = tid >> 2;   // 0..63
    const int seg = tid & 3;    // 4 x 16B per row
    async_copy16(gsrc + (size_t)row * D_MODEL + seg * 8, lbuf + row * 40 + seg * 8);
}

#define WMMA_F16(A, B, C) __builtin_amdgcn_wmma_f32_16x16x32_f16(false, A, false, B, (short)0, C, false, false)

// ---- fp32 -> fp16 convert ----
__global__ __launch_bounds__(256) void cvt_f32_f16(const float* __restrict__ src,
                                                   _Float16* __restrict__ dst, int n4) {
    int i = blockIdx.x * blockDim.x + threadIdx.x;
    if (i < n4) {
        float4 f = ((const float4*)src)[i];
        v4h h = { (_Float16)f.x, (_Float16)f.y, (_Float16)f.z, (_Float16)f.w };
        ((v4h*)dst)[i] = h;
    }
}

// ---- Fused QKV projection: out = X * W^T + b ----
// grid = (C/64, NTOK/256, 3); block = 256; wave owns 32x64 tile
__global__ __launch_bounds__(256) void qkv_gemm(
    const _Float16* __restrict__ Xh,
    const _Float16* __restrict__ Wqh, const _Float16* __restrict__ Wkh,
    const _Float16* __restrict__ Wvh,
    const float* __restrict__ bq, const float* __restrict__ bk, const float* __restrict__ bv,
    _Float16* __restrict__ Qh, _Float16* __restrict__ Kh, _Float16* __restrict__ Vt)
{
    __shared__ __align__(16) _Float16 Bs[2][64 * 40];
    const int tid = threadIdx.x;
    const int lane = tid & 31, wave = tid >> 5;
    const int lg = lane >> 4, ln = lane & 15;
    const int sel = blockIdx.z;
    const _Float16* W    = (sel == 0) ? Wqh : ((sel == 1) ? Wkh : Wvh);
    const float*    bias = (sel == 0) ? bq  : ((sel == 1) ? bk  : bv);
    const int M0 = blockIdx.y * 256 + wave * 32;
    const int N0 = blockIdx.x * 64;
    const _Float16* A0 = Xh + (size_t)M0 * D_MODEL;
    const _Float16* A1 = Xh + (size_t)(M0 + 16) * D_MODEL;
    const _Float16* WB = W + (size_t)N0 * D_MODEL;

    // prologue: stage chunk 0
    stage_b_chunk(WB, Bs[0], tid);
    wait_async_all();
    __syncthreads();

    v8f acc0[4] = {}, acc1[4] = {};
    v16h a0 = load_a_frag(A0, D_MODEL, lane);
    v16h a1 = load_a_frag(A1, D_MODEL, lane);
    int buf = 0;
    for (int kk = 0; kk < D_MODEL; kk += 32) {
        const bool last = (kk + 32 >= D_MODEL);
        if (!last) stage_b_chunk(WB + kk + 32, Bs[buf ^ 1], tid);
        v16h a0n = a0, a1n = a1;
        if (!last) {
            a0n = load_a_frag(A0 + kk + 32, D_MODEL, lane);
            a1n = load_a_frag(A1 + kk + 32, D_MODEL, lane);
        }
        #pragma unroll
        for (int j = 0; j < 4; ++j) {
            v16h bf = lds_b_frag(Bs[buf], j, lane);
            acc0[j] = WMMA_F16(a0, bf, acc0[j]);
            acc1[j] = WMMA_F16(a1, bf, acc1[j]);
        }
        wait_async_all();
        __syncthreads();
        a0 = a0n; a1 = a1n; buf ^= 1;
    }

    #pragma unroll
    for (int j = 0; j < 4; ++j) {
        const int o = N0 + j * 16 + ln;
        const float bo = bias[o];
        const int h = o >> 6, d = o & 63;
        #pragma unroll
        for (int sub = 0; sub < 2; ++sub) {
            const v8f* acc = sub ? acc1 : acc0;
            #pragma unroll
            for (int r = 0; r < 8; ++r) {
                const int n = M0 + sub * 16 + r + 8 * lg;
                const int b = n >> 11, t = n & (SEQ_T - 1);
                const float val = acc[j][r] + bo;
                if (sel == 2) {
                    Vt[(((size_t)(b * N_HEADS + h)) * D_HEAD + d) * SEQ_T + t] = (_Float16)val;
                } else {
                    _Float16* dst = (sel == 0) ? Qh : Kh;
                    dst[(((size_t)(b * N_HEADS + h)) * SEQ_T + t) * D_HEAD + d] = (_Float16)val;
                }
            }
        }
    }
}

// ---- Flash attention: grid = (T/128, B*H); block = 256; wave owns 16 q-rows ----
__global__ __launch_bounds__(256) void attn_kernel(
    const _Float16* __restrict__ Qh, const _Float16* __restrict__ Kh,
    const _Float16* __restrict__ Vt, _Float16* __restrict__ Ah)
{
    __shared__ __align__(16) _Float16 plds[8][16][80];  // per-wave 16x64 P tile
    const int lane = threadIdx.x & 31, wave = threadIdx.x >> 5;
    const int lg = lane >> 4, ln = lane & 15;
    const int bh = blockIdx.y;
    const int q0 = blockIdx.x * 128 + wave * 16;

    const _Float16* Qb = Qh + (size_t)bh * SEQ_T * D_HEAD + (size_t)q0 * D_HEAD;
    const _Float16* Kb = Kh + (size_t)bh * SEQ_T * D_HEAD;
    const _Float16* Vb = Vt + (size_t)bh * D_HEAD * SEQ_T;

    v16h qa[2];
    #pragma unroll
    for (int kc = 0; kc < 2; ++kc)
        qa[kc] = load_a_frag(Qb + kc * 32, D_HEAD, lane);

    float mrow[8], lrow[8];
    #pragma unroll
    for (int r = 0; r < 8; ++r) { mrow[r] = -INFINITY; lrow[r] = 0.f; }
    v8f oacc[4] = {};

    const float sc = 0.125f * 1.4426950408889634f;  // 1/sqrt(64) * log2(e)
    const int kmax = q0 + 15;
    for (int kb = 0; kb <= kmax; kb += 64) {
        if (kb + 64 <= kmax) {
            __builtin_prefetch(Kb + (size_t)(kb + 64) * D_HEAD, 0, 0);
            __builtin_prefetch(Vb + kb + 64, 0, 0);
        }
        // ---- S = Q * K^T : batch all 8 K-fragment loads, then 8 WMMAs ----
        v16h kf[4][2];
        #pragma unroll
        for (int j = 0; j < 4; ++j)
            #pragma unroll
            for (int kc = 0; kc < 2; ++kc)
                kf[j][kc] = load_b_frag(Kb + (size_t)(kb + j * 16) * D_HEAD + kc * 32, D_HEAD, lane);
        v8f s[4] = {};
        #pragma unroll
        for (int j = 0; j < 4; ++j)
            #pragma unroll
            for (int kc = 0; kc < 2; ++kc)
                s[j] = WMMA_F16(qa[kc], kf[j][kc], s[j]);

        // ---- mask + online softmax (row = 16 lanes of one half-wave) ----
        #pragma unroll
        for (int r = 0; r < 8; ++r) {
            const int q = q0 + r + 8 * lg;
            float mx = -INFINITY;
            #pragma unroll
            for (int j = 0; j < 4; ++j) {
                float v = s[j][r] * sc;
                v = ((kb + j * 16 + ln) <= q) ? v : -INFINITY;
                s[j][r] = v;
                mx = fmaxf(mx, v);
            }
            #pragma unroll
            for (int msk = 1; msk < 16; msk <<= 1)
                mx = fmaxf(mx, __shfl_xor(mx, msk, 32));
            const float mn    = fmaxf(mrow[r], mx);
            const float alpha = exp2f(mrow[r] - mn);
            mrow[r] = mn;
            float sum = 0.f;
            #pragma unroll
            for (int j = 0; j < 4; ++j) {
                const float p = exp2f(s[j][r] - mn);
                s[j][r] = p;
                sum += p;
            }
            #pragma unroll
            for (int msk = 1; msk < 16; msk <<= 1)
                sum += __shfl_xor(sum, msk, 32);
            lrow[r] = lrow[r] * alpha + sum;
            #pragma unroll
            for (int j = 0; j < 4; ++j)
                oacc[j][r] *= alpha;
            #pragma unroll
            for (int j = 0; j < 4; ++j)
                plds[wave][r + 8 * lg][j * 16 + ln] = (_Float16)s[j][r];
        }

        // ---- O += P * V : batch all V-fragment loads + P A-frags, then 8 WMMAs ----
        v16h vf[2][4];
        #pragma unroll
        for (int kc = 0; kc < 2; ++kc)
            #pragma unroll
            for (int j = 0; j < 4; ++j)
                vf[kc][j] = load_b_frag(Vb + (size_t)(j * 16) * SEQ_T + kb + kc * 32, SEQ_T, lane);
        v16h pa[2];
        #pragma unroll
        for (int kc = 0; kc < 2; ++kc)
            pa[kc] = load_a_frag(&plds[wave][0][0] + kc * 32, 80, lane);
        #pragma unroll
        for (int kc = 0; kc < 2; ++kc)
            #pragma unroll
            for (int j = 0; j < 4; ++j)
                oacc[j] = WMMA_F16(pa[kc], vf[kc][j], oacc[j]);
    }

    // normalize and scatter into [B,T,C] f16
    const int b = bh >> 4, h = bh & 15;
    #pragma unroll
    for (int r = 0; r < 8; ++r) {
        const float inv = 1.0f / lrow[r];
        const int q = q0 + r + 8 * lg;
        _Float16* dst = Ah + ((size_t)(b * SEQ_T + q)) * D_MODEL + h * D_HEAD;
        #pragma unroll
        for (int j = 0; j < 4; ++j)
            dst[j * 16 + ln] = (_Float16)(oacc[j][r] * inv);
    }
}

// ---- Output projection: out = A * Wo^T + bo (fp32 out) ----
// grid = (C/64, NTOK/256); block = 256; wave owns 32x64 tile
__global__ __launch_bounds__(256) void oproj_gemm(
    const _Float16* __restrict__ Ah, const _Float16* __restrict__ Woh,
    const float* __restrict__ bo, float* __restrict__ out)
{
    __shared__ __align__(16) _Float16 Bs[2][64 * 40];
    const int tid = threadIdx.x;
    const int lane = tid & 31, wave = tid >> 5;
    const int lg = lane >> 4, ln = lane & 15;
    const int M0 = blockIdx.y * 256 + wave * 32;
    const int N0 = blockIdx.x * 64;
    const _Float16* A0 = Ah + (size_t)M0 * D_MODEL;
    const _Float16* A1 = Ah + (size_t)(M0 + 16) * D_MODEL;
    const _Float16* WB = Woh + (size_t)N0 * D_MODEL;

    stage_b_chunk(WB, Bs[0], tid);
    wait_async_all();
    __syncthreads();

    v8f acc0[4] = {}, acc1[4] = {};
    v16h a0 = load_a_frag(A0, D_MODEL, lane);
    v16h a1 = load_a_frag(A1, D_MODEL, lane);
    int buf = 0;
    for (int kk = 0; kk < D_MODEL; kk += 32) {
        const bool last = (kk + 32 >= D_MODEL);
        if (!last) stage_b_chunk(WB + kk + 32, Bs[buf ^ 1], tid);
        v16h a0n = a0, a1n = a1;
        if (!last) {
            a0n = load_a_frag(A0 + kk + 32, D_MODEL, lane);
            a1n = load_a_frag(A1 + kk + 32, D_MODEL, lane);
        }
        #pragma unroll
        for (int j = 0; j < 4; ++j) {
            v16h bf = lds_b_frag(Bs[buf], j, lane);
            acc0[j] = WMMA_F16(a0, bf, acc0[j]);
            acc1[j] = WMMA_F16(a1, bf, acc1[j]);
        }
        wait_async_all();
        __syncthreads();
        a0 = a0n; a1 = a1n; buf ^= 1;
    }

    #pragma unroll
    for (int j = 0; j < 4; ++j) {
        const int o = N0 + j * 16 + ln;
        const float bb = bo[o];
        #pragma unroll
        for (int sub = 0; sub < 2; ++sub) {
            const v8f* acc = sub ? acc1 : acc0;
            #pragma unroll
            for (int r = 0; r < 8; ++r) {
                const int n = M0 + sub * 16 + r + 8 * lg;
                out[(size_t)n * D_MODEL + o] = acc[j][r] + bb;
            }
        }
    }
}

extern "C" void kernel_launch(void* const* d_in, const int* in_sizes, int n_in,
                              void* d_out, int out_size, void* d_ws, size_t ws_size,
                              hipStream_t stream) {
    const float* x  = (const float*)d_in[0];
    const float* Wq = (const float*)d_in[1];
    const float* bq = (const float*)d_in[2];
    const float* Wk = (const float*)d_in[3];
    const float* bk = (const float*)d_in[4];
    const float* Wv = (const float*)d_in[5];
    const float* bv = (const float*)d_in[6];
    const float* Wo = (const float*)d_in[7];
    const float* bo = (const float*)d_in[8];
    float* out = (float*)d_out;

    char* ws = (char*)d_ws;
    _Float16* Xh  = (_Float16*)(ws);                            // 8 MB  [4096,1024]
    _Float16* Wqh = (_Float16*)(ws + ((size_t)8  << 20));       // 2 MB
    _Float16* Wkh = (_Float16*)(ws + ((size_t)10 << 20));       // 2 MB
    _Float16* Wvh = (_Float16*)(ws + ((size_t)12 << 20));       // 2 MB
    _Float16* Woh = (_Float16*)(ws + ((size_t)14 << 20));       // 2 MB
    _Float16* Qh  = (_Float16*)(ws + ((size_t)16 << 20));       // 8 MB  [B,H,T,Dh]
    _Float16* Kh  = (_Float16*)(ws + ((size_t)24 << 20));       // 8 MB  [B,H,T,Dh]
    _Float16* Vt  = (_Float16*)(ws + ((size_t)32 << 20));       // 8 MB  [B,H,Dh,T]
    _Float16* Ah  = (_Float16*)(ws + ((size_t)40 << 20));       // 8 MB  [B,T,C]

    cvt_f32_f16<<<(NTOK * D_MODEL / 4) / 256, 256, 0, stream>>>(x,  Xh,  NTOK * D_MODEL / 4);
    cvt_f32_f16<<<(D_MODEL * D_MODEL / 4) / 256, 256, 0, stream>>>(Wq, Wqh, D_MODEL * D_MODEL / 4);
    cvt_f32_f16<<<(D_MODEL * D_MODEL / 4) / 256, 256, 0, stream>>>(Wk, Wkh, D_MODEL * D_MODEL / 4);
    cvt_f32_f16<<<(D_MODEL * D_MODEL / 4) / 256, 256, 0, stream>>>(Wv, Wvh, D_MODEL * D_MODEL / 4);
    cvt_f32_f16<<<(D_MODEL * D_MODEL / 4) / 256, 256, 0, stream>>>(Wo, Woh, D_MODEL * D_MODEL / 4);

    qkv_gemm<<<dim3(D_MODEL / 64, NTOK / 256, 3), 256, 0, stream>>>(
        Xh, Wqh, Wkh, Wvh, bq, bk, bv, Qh, Kh, Vt);

    attn_kernel<<<dim3(SEQ_T / 128, BATCH * N_HEADS), 256, 0, stream>>>(Qh, Kh, Vt, Ah);

    oproj_gemm<<<dim3(D_MODEL / 64, NTOK / 256), 256, 0, stream>>>(Ah, Woh, bo, out);
}
